// AttentionBlock_14482629722782
// MI455X (gfx1250) — matrix operand order repaired and checked by reading, not compiled, not run on hardware
//
#include <hip/hip_runtime.h>
#include <hip/hip_bf16.h>
#include <stdint.h>

// ---------------------------------------------------------------------------
// Problem constants (from reference): B=8, S=2048, D=E=128
// ---------------------------------------------------------------------------
#define BATCH 8
#define SEQ   2048
#define DIM   128

typedef __attribute__((ext_vector_type(16))) __bf16   bf16x16;
typedef __attribute__((ext_vector_type(2)))  __bf16   bf16x2;
typedef __attribute__((ext_vector_type(8)))  float    f32x8;
typedef __attribute__((ext_vector_type(4)))  uint32_t u32x4;
typedef __attribute__((ext_vector_type(2)))  uint32_t u32x2;

union Frag {
    bf16x16  h;
    u32x4    q[2];
    uint32_t u[8];
};

__device__ __forceinline__ uint16_t f2bf(float f) {
    uint32_t u = __builtin_bit_cast(uint32_t, f);
    u += 0x7FFFu + ((u >> 16) & 1u);          // round-to-nearest-even
    return (uint16_t)(u >> 16);
}

__device__ __forceinline__ uint32_t pack_bf16(float a, float b) {
#if defined(__has_builtin)
#if __has_builtin(__builtin_amdgcn_cvt_pk_bf16_f32)
    bf16x2 v = __builtin_amdgcn_cvt_pk_bf16_f32(a, b);
    return __builtin_bit_cast(uint32_t, v);
#else
    return (uint32_t)f2bf(a) | ((uint32_t)f2bf(b) << 16);
#endif
#else
    return (uint32_t)f2bf(a) | ((uint32_t)f2bf(b) << 16);
#endif
}

#define WMMA_BF16(A, B, C) \
    __builtin_amdgcn_wmma_f32_16x16x32_bf16(false, (A), false, (B), (short)0, (C), false, false)

// ---------------------------------------------------------------------------
// Kernel 0: W [D][E] f32 -> Wt [E][D] bf16 (transposed, B-fragment friendly)
// grid = (E, 2), block = D
// ---------------------------------------------------------------------------
__global__ void wprep_bf16_kernel(const float* __restrict__ Wq,
                                  const float* __restrict__ Wk,
                                  uint16_t* __restrict__ Wqt,
                                  uint16_t* __restrict__ Wkt) {
    const int e = blockIdx.x;
    const int d = threadIdx.x;
    const float* W = (blockIdx.y == 0) ? Wq : Wk;
    uint16_t*   Wt = (blockIdx.y == 0) ? Wqt : Wkt;
    Wt[(size_t)e * DIM + d] = f2bf(W[(size_t)d * DIM + e]);
}

// ---------------------------------------------------------------------------
// Kernel 1: X [B][S][D] f32 -> Xt [B][D][S] bf16 (LDS-tiled transpose)
//                           -> Xr [B][S][D] bf16 (row-major copy)
// ---------------------------------------------------------------------------
__global__ void xprep_bf16_kernel(const float* __restrict__ X,
                                  uint16_t* __restrict__ Xt,
                                  uint16_t* __restrict__ Xr) {
    __shared__ float tile[32][33];
    const int b  = blockIdx.z;
    const int s0 = blockIdx.x * 32;
    const int d0 = blockIdx.y * 32;
    const int tx = threadIdx.x & 31;
    const int ty = threadIdx.x >> 5;          // 0..7

#pragma unroll
    for (int k = 0; k < 4; ++k) {
        int s = ty + 8 * k;
        float v = X[((size_t)b * SEQ + s0 + s) * DIM + d0 + tx];
        tile[s][tx] = v;
        Xr[((size_t)b * SEQ + s0 + s) * DIM + d0 + tx] = f2bf(v);   // coalesced
    }
    __syncthreads();
#pragma unroll
    for (int k = 0; k < 4; ++k) {
        int d = ty + 8 * k;
        Xt[((size_t)b * DIM + d0 + d) * SEQ + s0 + tx] = f2bf(tile[tx][d]);
    }
}

// ---------------------------------------------------------------------------
// Kernel 2: Q = X@Wq + bq, K = X@Wk + bk  via bf16 WMMA, f32 accumulate.
// All operands pre-converted to bf16 -> inner loop is b128 loads + WMMA only.
// One wave computes a 16-row x 128-col tile of one projection.
// grid = (16384/16, 2), block = 32
// ---------------------------------------------------------------------------
__global__ void proj_bf16_wmma_kernel(const uint16_t* __restrict__ Xr,
                                      const uint16_t* __restrict__ Wqt,
                                      const float*    __restrict__ bq,
                                      const uint16_t* __restrict__ Wkt,
                                      const float*    __restrict__ bk,
                                      uint16_t* __restrict__ Qb,
                                      uint16_t* __restrict__ Kb) {
    const int lane = threadIdx.x;
    const int hf   = lane >> 4;               // lane half (0/1)
    const int lc   = lane & 15;
    const int r0   = blockIdx.x * 16;

    const uint16_t* Wt  = (blockIdx.y == 0) ? Wqt : Wkt;
    const float*    bia = (blockIdx.y == 0) ? bq  : bk;
    uint16_t*       Y   = (blockIdx.y == 0) ? Qb  : Kb;

    f32x8 acc[8];
    const f32x8 fz = {0.f, 0.f, 0.f, 0.f, 0.f, 0.f, 0.f, 0.f};
#pragma unroll
    for (int t = 0; t < 8; ++t) acc[t] = fz;

    const uint16_t* xp = Xr + (size_t)(r0 + lc) * DIM;   // A row for this lane

#pragma unroll
    for (int ks = 0; ks < 4; ++ks) {
        // A fragment: lane row lc, V0-3 = feats ks*32 + 8*hf + {0..7}, V4-7 = +16
        Frag a;
        a.q[0] = *(const u32x4*)(xp + ks * 32 + hf * 8);
        a.q[1] = *(const u32x4*)(xp + ks * 32 + hf * 8 + 16);
#pragma unroll
        for (int t = 0; t < 8; ++t) {
            // B fragment: column e = 16t+lc, rows k = ks*32 + 8*hf + {0..7}, +16
            Frag wb;
            const uint16_t* wp = Wt + (size_t)(t * 16 + lc) * DIM + ks * 32 + hf * 8;
            wb.q[0] = *(const u32x4*)(wp);
            wb.q[1] = *(const u32x4*)(wp + 16);
            acc[t] = WMMA_BF16(a.h, wb.h, acc[t]);
        }
    }
    // C/D layout: VGPR i -> row (hf*8 + i), column lc (within 16-wide N tile t)
#pragma unroll
    for (int t = 0; t < 8; ++t) {
        float bv = bia[t * 16 + lc];
#pragma unroll
        for (int i = 0; i < 8; ++i) {
            Y[(size_t)(r0 + hf * 8 + i) * DIM + t * 16 + lc] = f2bf(acc[t][i] + bv);
        }
    }
}

// ---------------------------------------------------------------------------
// Kernel 3: fused flash attention (no score materialization).
// One wave per 16 query rows; 32 chunks of 64 keys.
// Key interleave: score tile t (t=0..3), column c <-> key ch*64 + 4c + t, so
// the 4 probabilities a lane makes per row are ADJACENT keys -> packed dword
// conversion + one ds_store_b64 per row, and the A-fragment read-back from
// LDS is in natural key order.
//   S  = Q(16x128).K^T(128x64) -> four 16x16 f32 tiles     (16 WMMA / chunk)
//   online softmax (row max/sum via __shfl_xor within 16-lane halves)
//   O += P(16x64).X(64x128)                                (16 WMMA / chunk)
// grid = (S/16, B), block = 32
// ---------------------------------------------------------------------------
__global__ void flash_attn_kernel(const uint16_t* __restrict__ Qb,
                                  const uint16_t* __restrict__ Kb,
                                  const uint16_t* __restrict__ Xt,
                                  float* __restrict__ out) {
    __shared__ uint16_t ldsP[16 * 64];        // 2 KB: P tile 16 rows x 64 keys
    const int lane = threadIdx.x;
    const int hf   = lane >> 4;
    const int lc   = lane & 15;
    const int b    = blockIdx.y;
    const int q0   = blockIdx.x * 16;

    // Preload Q A-fragments (4 k-steps of 32 features each)
    Frag qf[4];
    {
        const uint16_t* qp = Qb + ((size_t)b * SEQ + q0 + lc) * DIM;
#pragma unroll
        for (int ks = 0; ks < 4; ++ks) {
            qf[ks].q[0] = *(const u32x4*)(qp + ks * 32 + hf * 8);
            qf[ks].q[1] = *(const u32x4*)(qp + ks * 32 + hf * 8 + 16);
        }
    }

    const f32x8 fz = {0.f, 0.f, 0.f, 0.f, 0.f, 0.f, 0.f, 0.f};
    f32x8 O[8];
#pragma unroll
    for (int j = 0; j < 8; ++j) O[j] = fz;
    float m[8], l[8];
#pragma unroll
    for (int i = 0; i < 8; ++i) { m[i] = -__builtin_inff(); l[i] = 0.f; }

    const uint16_t* kbase = Kb + (size_t)b * SEQ * DIM;
    const uint16_t* xbase = Xt + (size_t)b * DIM * SEQ;

    for (int ch = 0; ch < SEQ / 64; ++ch) {
        // ---- prefetch next chunk (emits global_prefetch_b8) ----
        if (ch + 1 < SEQ / 64) {
            __builtin_prefetch(kbase + (size_t)((ch + 1) * 64 + lane * 2) * DIM, 0, 1);
            __builtin_prefetch(kbase + (size_t)((ch + 1) * 64 + lane * 2 + 1) * DIM, 0, 1);
            __builtin_prefetch(xbase + (size_t)(lane * 4 + hf * 16) * SEQ + (ch + 1) * 64, 0, 1);
        }

        // ---- scores: four 16x16 tiles, tile t col c = key ch*64 + 4c + t ----
        f32x8 s[4] = {fz, fz, fz, fz};
        // B-fragment row for this lane in tile t: key ch*64 + 4*lc + t
        const uint16_t* krow = kbase + (size_t)(ch * 64 + 4 * lc) * DIM;
#pragma unroll
        for (int ks = 0; ks < 4; ++ks) {
#pragma unroll
            for (int t = 0; t < 4; ++t) {
                Frag kb;
                const uint16_t* kp = krow + (size_t)t * DIM + ks * 32 + hf * 8;
                kb.q[0] = *(const u32x4*)(kp);
                kb.q[1] = *(const u32x4*)(kp + 16);
                s[t] = WMMA_BF16(qf[ks].h, kb.h, s[t]);
            }
        }

        // ---- online softmax update (row i lives in lanes of this half) ----
        float alpha[8];
#pragma unroll
        for (int i = 0; i < 8; ++i) {
            float v = fmaxf(fmaxf(s[0][i], s[1][i]), fmaxf(s[2][i], s[3][i]));
            v = fmaxf(v, __shfl_xor(v, 1, 32));
            v = fmaxf(v, __shfl_xor(v, 2, 32));
            v = fmaxf(v, __shfl_xor(v, 4, 32));
            v = fmaxf(v, __shfl_xor(v, 8, 32));
            float mn = fmaxf(m[i], v);
            alpha[i] = __expf(m[i] - mn);
            m[i] = mn;
            float p0 = __expf(s[0][i] - mn);
            float p1 = __expf(s[1][i] - mn);
            float p2 = __expf(s[2][i] - mn);
            float p3 = __expf(s[3][i] - mn);
            float rs = (p0 + p1) + (p2 + p3);
            rs += __shfl_xor(rs, 1, 32);
            rs += __shfl_xor(rs, 2, 32);
            rs += __shfl_xor(rs, 4, 32);
            rs += __shfl_xor(rs, 8, 32);
            l[i] = l[i] * alpha[i] + rs;
            // keys 4*lc..4*lc+3 are adjacent -> packed b64 store
            u32x2 w = { pack_bf16(p0, p1), pack_bf16(p2, p3) };
            *(u32x2*)(ldsP + (hf * 8 + i) * 64 + 4 * lc) = w;
        }
#pragma unroll
        for (int j = 0; j < 8; ++j)
#pragma unroll
            for (int i = 0; i < 8; ++i) O[j][i] *= alpha[i];

        // ---- P re-layout: LDS -> two A fragments (keys 0..31 / 32..63) ----
        asm volatile("s_wait_dscnt 0x0" ::: "memory");
        Frag pf0, pf1;
        pf0.q[0] = *(const u32x4*)(ldsP + lc * 64 + hf * 8);
        pf0.q[1] = *(const u32x4*)(ldsP + lc * 64 + hf * 8 + 16);
        pf1.q[0] = *(const u32x4*)(ldsP + lc * 64 + 32 + hf * 8);
        pf1.q[1] = *(const u32x4*)(ldsP + lc * 64 + 32 + hf * 8 + 16);

        // ---- O += P . X  (Xt gives contiguous per-lane B-fragment loads) ----
#pragma unroll
        for (int j = 0; j < 8; ++j) {
            Frag xb0, xb1;
            const uint16_t* xp = xbase + (size_t)(16 * j + lc) * SEQ + ch * 64 + hf * 8;
            xb0.q[0] = *(const u32x4*)(xp);
            xb0.q[1] = *(const u32x4*)(xp + 16);
            xb1.q[0] = *(const u32x4*)(xp + 32);
            xb1.q[1] = *(const u32x4*)(xp + 48);
            O[j] = WMMA_BF16(pf0.h, xb0.h, O[j]);
            O[j] = WMMA_BF16(pf1.h, xb1.h, O[j]);
        }
    }

    // ---- finalize: O / l ----
#pragma unroll
    for (int i = 0; i < 8; ++i) {
        float inv = 1.0f / l[i];
        float* op = out + ((size_t)b * SEQ + q0 + hf * 8 + i) * DIM + lc;
#pragma unroll
        for (int j = 0; j < 8; ++j) op[16 * j] = O[j][i] * inv;
    }
}

// ---------------------------------------------------------------------------
// Host launch
// ---------------------------------------------------------------------------
extern "C" void kernel_launch(void* const* d_in, const int* in_sizes, int n_in,
                              void* d_out, int out_size, void* d_ws, size_t ws_size,
                              hipStream_t stream) {
    const float* X  = (const float*)d_in[0];
    const float* Wq = (const float*)d_in[1];
    const float* bq = (const float*)d_in[2];
    const float* Wk = (const float*)d_in[3];
    const float* bk = (const float*)d_in[4];
    float* out = (float*)d_out;

    // workspace: Qb | Kb | Xt | Xr (4MB each, bf16) | Wqt | Wkt (32KB each)
    uint16_t* Qb  = (uint16_t*)d_ws;
    uint16_t* Kb  = Qb  + (size_t)BATCH * SEQ * DIM;
    uint16_t* Xt  = Kb  + (size_t)BATCH * SEQ * DIM;
    uint16_t* Xr  = Xt  + (size_t)BATCH * SEQ * DIM;
    uint16_t* Wqt = Xr  + (size_t)BATCH * SEQ * DIM;
    uint16_t* Wkt = Wqt + (size_t)DIM * DIM;

    wprep_bf16_kernel<<<dim3(DIM, 2), DIM, 0, stream>>>(Wq, Wk, Wqt, Wkt);
    xprep_bf16_kernel<<<dim3(SEQ / 32, DIM / 32, BATCH), 256, 0, stream>>>(X, Xt, Xr);
    proj_bf16_wmma_kernel<<<dim3((BATCH * SEQ) / 16, 2), 32, 0, stream>>>(
        Xr, Wqt, bq, Wkt, bk, Qb, Kb);
    flash_attn_kernel<<<dim3(SEQ / 16, BATCH), 32, 0, stream>>>(Qb, Kb, Xt, out);
}
